// TransformerModel_65197603553962
// MI455X (gfx1250) — matrix operand order repaired and checked by reading, compiled, not run on hardware
//
#include <hip/hip_runtime.h>
#include <hip/hip_bf16.h>
#include <math.h>

typedef _Float16 half_t;
typedef __attribute__((ext_vector_type(16))) _Float16 v16h;
typedef __attribute__((ext_vector_type(8)))  _Float16 v8h;
typedef __attribute__((ext_vector_type(8)))  float    v8f;

static constexpr int T_B  = 256;
static constexpr int T_S  = 200;
static constexpr int T_D  = 300;
static constexpr int T_DP = 320;   // D padded to K-multiple-of-32
static constexpr int T_H  = 6;
static constexpr int T_DH = 50;
static constexpr int T_SP = 224;   // S padded to multiple of 32 (attention K dim)
static constexpr int T_M  = T_B * T_S;   // 51200 rows, multiple of 64

// ---------------------------------------------------------------------------
// CDNA5 async global->LDS copy (ASYNCcnt-tracked DMA, no VGPR round trip)
// lds_off: byte offset inside this workgroup's (dynamic) LDS allocation.
// ---------------------------------------------------------------------------
__device__ __forceinline__ void async_ld16(unsigned lds_off, const void* gaddr) {
  asm volatile("global_load_async_to_lds_b128 %0, %1, off"
               :: "v"(lds_off), "v"((unsigned long long)(uintptr_t)gaddr)
               : "memory");
}
__device__ __forceinline__ void async_wait0() {
  asm volatile("s_wait_asynccnt 0x0" ::: "memory");
}

// ---------------------------------------------------------------------------
// WMMA helpers (CDNA5 wave32, V_WMMA_F32_16X16X32_F16)
// ---------------------------------------------------------------------------
__device__ __forceinline__ v8f wmma16(v16h a, v16h b, v8f c) {
  return __builtin_amdgcn_wmma_f32_16x16x32_f16(false, a, false, b, (short)0, c,
                                                false, false);
}

__device__ __forceinline__ float gelu_f(float x) {
  return 0.5f * x * (1.0f + tanhf(0.7978845608028654f * (x + 0.044715f * x * x * x)));
}

// A fragment (16x32 f16, M x K), row-major source with leading dim ld (halves).
// ISA layout: lane m=lane%16; lanes 0-15 hold K {0..7,16..23}, lanes 16-31 {8..15,24..31}.
__device__ __forceinline__ v16h frag_a_h(const half_t* p, int ld, int lane) {
  int m = lane & 15, hs = lane >> 4;
  const v8h* q0 = (const v8h*)(p + m * ld + hs * 8);
  const v8h* q1 = (const v8h*)(p + m * ld + 16 + hs * 8);
  v8h a = *q0, b = *q1;
  v16h r;
#pragma unroll
  for (int e = 0; e < 8; e++) { r[e] = a[e]; r[e + 8] = b[e]; }
  return r;
}

// B fragment (32x16 f16, K x N) from N-major source, ld = K stride (halves).
// Lane column n=lane%16; lanes 0-15 hold K {0..7,16..23}, lanes 16-31 {8..15,24..31}.
__device__ __forceinline__ v16h frag_b_t(const half_t* p, int ld, int lane) {
  int n = lane & 15, ks = (lane >> 4) * 8;
  const v8h* q = (const v8h*)(p + n * ld + ks);
  v8h a = q[0], b = q[2];
  v16h r;
#pragma unroll
  for (int e = 0; e < 8; e++) { r[e] = a[e]; r[e + 8] = b[e]; }
  return r;
}

// A fragment from f32 row-major source (softmax probabilities), convert to f16.
__device__ __forceinline__ v16h frag_a_f32(const float* p, int ld, int k0, int lane) {
  int m = lane & 15, hs = lane >> 4;
  const float* r0 = p + m * ld + k0 + hs * 8;
  v16h r;
#pragma unroll
  for (int e = 0; e < 8; e++) { r[e] = (half_t)r0[e]; r[e + 8] = (half_t)r0[16 + e]; }
  return r;
}

// ---------------------------------------------------------------------------
// Weight pad + transpose: fp32 [L][K][N] -> f16 [L][Np][Kp] (N-major), zero fill.
// Weights are reused by 800 M-blocks, so transposing once here makes every
// GEMM LDS stage a contiguous b128 copy (no ds_store_b16 scatter).
// ---------------------------------------------------------------------------
__global__ void k_pad_w(const float* __restrict__ src, half_t* __restrict__ dst,
                        int K, int N, int Kp, int Np, int L) {
  int idx = blockIdx.x * blockDim.x + threadIdx.x;
  int tot = L * Kp * Np;
  if (idx >= tot) return;
  int k = idx % Kp;
  int n = (idx / Kp) % Np;
  int l = idx / (Kp * Np);
  float v = 0.0f;
  if (k < K && n < N) v = src[((size_t)l * K + k) * N + n];
  dst[idx] = (half_t)v;
}

__global__ void k_zero_attn_pad(half_t* __restrict__ attn16) {
  int idx = blockIdx.x * blockDim.x + threadIdx.x;
  if (idx >= T_M * (T_DP - T_D)) return;
  int m = idx / (T_DP - T_D), j = idx % (T_DP - T_D);
  attn16[(size_t)m * T_DP + T_D + j] = (half_t)0.0f;
}

// ---------------------------------------------------------------------------
// Embedding gather + sinusoidal positional encoding + token pad mask
// ---------------------------------------------------------------------------
__global__ void k_embed(const int* __restrict__ tok, const float* __restrict__ emb,
                        float* __restrict__ h, half_t* __restrict__ h16,
                        unsigned char* __restrict__ mask) {
  int m = blockIdx.x * blockDim.x + threadIdx.x;
  if (m >= T_M) return;
  int s = m % T_S;
  int t = tok[m];
  int pad = (t == 0);
  mask[m] = (unsigned char)pad;
  const float* e    = emb + (size_t)t * T_D;
  float*       hr   = h   + (size_t)m * T_D;
  half_t*      h16r = h16 + (size_t)m * T_DP;
  for (int c = 0; c < T_D; c++) {
    float pe = 0.0f;
    if (!pad) {
      float ex  = (float)(2 * (c >> 1)) * (1.0f / (float)T_D);
      float ang = (float)s * powf(10000.0f, -ex);
      pe = (c & 1) ? cosf(ang) : sinf(ang);
    }
    float v = e[c] + pe;
    hr[c] = v;
    h16r[c] = (half_t)v;
  }
  for (int c = T_D; c < T_DP; c++) h16r[c] = (half_t)0.0f;
}

__global__ void k_allpad(const unsigned char* __restrict__ mask,
                         unsigned char* __restrict__ allpad) {
  int b = threadIdx.x;
  if (b >= T_B) return;
  int ap = 1;
  for (int s = 0; s < T_S; s++) ap &= (int)mask[b * T_S + s];
  allpad[b] = (unsigned char)ap;
}

// ---------------------------------------------------------------------------
// Generic WMMA GEMM: C[M][Npad] = A[M][Kp] (f16) x BwT[Npad][Kp] (f16, N-major)
// 128 threads = 4 waves, 64x64 C tile/block. Tiles staged to LDS with
// GLOBAL_LOAD_ASYNC_TO_LDS_B128 (dynamic LDS: As @0, Bs @4096, 8KB total).
// ---------------------------------------------------------------------------
__global__ void __launch_bounds__(128)
k_gemm(const half_t* __restrict__ A, const half_t* __restrict__ BwT,
       const float* __restrict__ bias, int nreal, int Kp, int Npad,
       float* __restrict__ C32, half_t* __restrict__ C16, int act_gelu) {
  extern __shared__ char smem[];
  half_t* As = (half_t*)smem;            // [64][32], ld=32
  half_t* Bs = (half_t*)(smem + 4096);   // [64][32] N-major, ld=32

  int n0 = blockIdx.x * 64;
  int m0 = blockIdx.y * 64;
  int tid = threadIdx.x, lane = tid & 31, wav = tid >> 5;
  int wr = wav >> 1, wc = wav & 1;
  int row = tid >> 1, seg = tid & 1;     // staging role: 64 rows x 2 half-rows

  v8f acc[2][2] = {};

  for (int k0 = 0; k0 < Kp; k0 += 32) {
    // A tile 64x32: contiguous 32B per thread -> 2 async b128
    {
      const half_t* g = A + (size_t)(m0 + row) * Kp + k0 + seg * 16;
      unsigned lo = (unsigned)(row * 64 + seg * 32);
      async_ld16(lo, g);
      async_ld16(lo + 16, g + 8);
      if (k0 + 32 < Kp) __builtin_prefetch(g + 32, 0, 1);
    }
    // B tile (N-major) 64x32: contiguous 32B per thread -> 2 async b128
    {
      const half_t* g = BwT + (size_t)(n0 + row) * Kp + k0 + seg * 16;
      unsigned lo = (unsigned)(4096 + row * 64 + seg * 32);
      async_ld16(lo, g);
      async_ld16(lo + 16, g + 8);
    }
    async_wait0();
    __syncthreads();
    v16h af0 = frag_a_h(As + (wr * 32) * 32,      32, lane);
    v16h af1 = frag_a_h(As + (wr * 32 + 16) * 32, 32, lane);
    v16h bf0 = frag_b_t(Bs + (wc * 32) * 32,      32, lane);
    v16h bf1 = frag_b_t(Bs + (wc * 32 + 16) * 32, 32, lane);
    acc[0][0] = wmma16(af0, bf0, acc[0][0]);
    acc[0][1] = wmma16(af0, bf1, acc[0][1]);
    acc[1][0] = wmma16(af1, bf0, acc[1][0]);
    acc[1][1] = wmma16(af1, bf1, acc[1][1]);
    __syncthreads();
  }

  int n  = lane & 15;
  int mo = (lane >> 4) * 8;
#pragma unroll
  for (int i = 0; i < 2; i++) {
#pragma unroll
    for (int j = 0; j < 2; j++) {
      int col  = n0 + wc * 32 + j * 16 + n;
      float bv = (col < nreal) ? bias[col] : 0.0f;
      int rowb = m0 + wr * 32 + i * 16 + mo;
#pragma unroll
      for (int r = 0; r < 8; r++) {
        float v = acc[i][j][r] + bv;
        if (act_gelu) v = gelu_f(v);
        if (C32) C32[(size_t)(rowb + r) * Npad + col] = v;
        if (C16) C16[(size_t)(rowb + r) * Npad + col] = (half_t)v;
      }
    }
  }
}

// ---------------------------------------------------------------------------
// Repack QKV GEMM output into attention staging (q scaled by 1/sqrt(DH)):
//   qp [B][H][SP][64]  (q rows, A-operand layout)
//   kp [B][H][SP][64]  (k rows; N-major for the score GEMM's B operand)
//   vtp[B][H][64][SP]  (v transposed; N-major for the p@v GEMM's B operand)
// ---------------------------------------------------------------------------
__global__ void k_qkv_pack(const half_t* __restrict__ qkv16,
                           half_t* __restrict__ qp, half_t* __restrict__ kp,
                           half_t* __restrict__ vtp) {
  int idx = blockIdx.x * blockDim.x + threadIdx.x;
  if (idx >= T_B * T_H * T_SP * 64) return;
  int d  = idx & 63;
  int s  = (idx >> 6) % T_SP;
  int bh = idx / (T_SP * 64);
  int hh = bh % T_H;
  int b  = bh / T_H;
  half_t q = (half_t)0.0f, k = (half_t)0.0f, v = (half_t)0.0f;
  if (s < T_S && d < T_DH) {
    const half_t* row = qkv16 + (size_t)(b * T_S + s) * 960;
    q = (half_t)((float)row[hh * T_DH + d] * 0.14142135623730950488f); // 1/sqrt(50)
    k = row[300 + hh * T_DH + d];
    v = row[600 + hh * T_DH + d];
  }
  qp[idx] = q;
  kp[idx] = k;
  vtp[(size_t)bh * 64 * T_SP + d * T_SP + s] = v;
}

// ---------------------------------------------------------------------------
// Fused attention: scores = q k^T (WMMA) -> masked softmax (LDS) -> p v (WMMA)
// One block per (q-tile of 64 rows, head, batch). Dynamic LDS 112KB (<320KB/WGP):
//   kt @0 (28672B, [SP][64]), vt @28672 ([64][SP]), sc @57344 (f32 [64][SP]).
// k/v tiles staged via async global->LDS DMA.
// ---------------------------------------------------------------------------
__global__ void __launch_bounds__(128)
k_attn(const half_t* __restrict__ qp, const half_t* __restrict__ kp,
       const half_t* __restrict__ vtp, const unsigned char* __restrict__ mask,
       const unsigned char* __restrict__ allpad, half_t* __restrict__ attn16) {
  extern __shared__ char smem[];
  half_t* kt = (half_t*)smem;                  // [SP][64]
  half_t* vt = (half_t*)(smem + 28672);        // [64][SP]
  float*  sc = (float*)(smem + 57344);         // [64][SP]

  int qt = blockIdx.x, hh = blockIdx.y, b = blockIdx.z;
  int bh = b * T_H + hh;
  int tid = threadIdx.x, lane = tid & 31, wav = tid >> 5;

  const char* gk = (const char*)(kp  + (size_t)bh * T_SP * 64);
  const char* gv = (const char*)(vtp + (size_t)bh * 64 * T_SP);
  const half_t* qg = qp + (size_t)bh * T_SP * 64;

  // contiguous async DMA: 28672B each, 16B chunks
  for (int i = tid * 16; i < T_SP * 64 * 2; i += 128 * 16) {
    async_ld16((unsigned)i, gk + i);
    async_ld16((unsigned)(28672 + i), gv + i);
  }
  async_wait0();
  __syncthreads();

  int m = lane & 15, hs = lane >> 4;

  // q fragments for this wave's 16 rows (K = 64 -> two frags), from global
  v16h af0 = {}, af1 = {};
  {
    int rq = qt * 64 + wav * 16 + m;
    if (rq < T_SP) {
      const half_t* p = qg + (size_t)rq * 64;
      v8h x0 = *(const v8h*)(p + hs * 8);
      v8h x1 = *(const v8h*)(p + 16 + hs * 8);
      v8h x2 = *(const v8h*)(p + 32 + hs * 8);
      v8h x3 = *(const v8h*)(p + 48 + hs * 8);
#pragma unroll
      for (int e = 0; e < 8; e++) {
        af0[e] = x0[e]; af0[e + 8] = x1[e];
        af1[e] = x2[e]; af1[e + 8] = x3[e];
      }
    }
  }

  // scores: wave w -> local rows [w*16, w*16+16), all 14 column tiles
  for (int ct = 0; ct < T_SP / 16; ct++) {
    v8f acc = {};
    v16h b0 = frag_b_t(kt + (ct * 16) * 64,      64, lane);
    v16h b1 = frag_b_t(kt + (ct * 16) * 64 + 32, 64, lane);
    acc = wmma16(af0, b0, acc);
    acc = wmma16(af1, b1, acc);
    int col  = ct * 16 + m;
    bool msk = (col >= T_S) || (mask[b * T_S + col] != 0);
    int rl   = wav * 16 + hs * 8;
#pragma unroll
    for (int r = 0; r < 8; r++)
      sc[(rl + r) * T_SP + col] = msk ? -__builtin_inff() : acc[r];
  }
  __syncthreads();

  // masked softmax, in place (one thread per row)
  if (tid < 64) {
    float* row = sc + tid * T_SP;
    if (allpad[b]) {
      for (int c = 0; c < T_SP; c++) row[c] = 0.0f;
    } else {
      float mx = -__builtin_inff();
      for (int c = 0; c < T_SP; c++) mx = fmaxf(mx, row[c]);
      float sm = 0.0f;
      for (int c = 0; c < T_SP; c++) { float e = __expf(row[c] - mx); row[c] = e; sm += e; }
      float inv = 1.0f / sm;
      for (int c = 0; c < T_SP; c++) row[c] *= inv;
    }
  }
  __syncthreads();

  // o = p @ v : [16 x SP] x [SP x 64] per wave
  v8f o[4] = {};
  for (int k0 = 0; k0 < T_SP; k0 += 32) {
    v16h pa = frag_a_f32(sc + (wav * 16) * T_SP, T_SP, k0, lane);
#pragma unroll
    for (int j = 0; j < 4; j++) {
      v16h vb = frag_b_t(vt + (j * 16) * T_SP + k0, T_SP, lane);
      o[j] = wmma16(pa, vb, o[j]);
    }
  }

  // scatter into attention-output staging [M][DP] at head column offset
#pragma unroll
  for (int j = 0; j < 4; j++) {
    int d = j * 16 + m;
    if (d < T_DH) {
      int rl = wav * 16 + hs * 8;
#pragma unroll
      for (int r = 0; r < 8; r++) {
        int srow = qt * 64 + rl + r;
        if (srow < T_S)
          attn16[(size_t)(b * T_S + srow) * T_DP + hh * T_DH + d] = (half_t)o[j][r];
      }
    }
  }
}

// ---------------------------------------------------------------------------
// Residual + LayerNorm (one wave per row) + recompute all(h==0) mask
// ---------------------------------------------------------------------------
__global__ void __launch_bounds__(32)
k_add_ln(const float* __restrict__ hin, const float* __restrict__ delta,
         const float* __restrict__ lns, const float* __restrict__ lnb,
         float* __restrict__ hout, half_t* __restrict__ h16,
         unsigned char* __restrict__ mask) {
  int mrow = blockIdx.x;
  int lane = threadIdx.x;
  float vals[10];
  float sum = 0.0f;
#pragma unroll
  for (int i = 0; i < 10; i++) {
    int c = lane + i * 32;
    float x = 0.0f;
    if (c < T_D) x = hin[(size_t)mrow * T_D + c] + delta[(size_t)mrow * T_DP + c];
    vals[i] = x;
    sum += x;
  }
  for (int o = 16; o > 0; o >>= 1) sum += __shfl_xor(sum, o, 32);
  float mean = sum * (1.0f / (float)T_D);
  float vs = 0.0f;
#pragma unroll
  for (int i = 0; i < 10; i++) {
    int c = lane + i * 32;
    if (c < T_D) { float d = vals[i] - mean; vs += d * d; }
  }
  for (int o = 16; o > 0; o >>= 1) vs += __shfl_xor(vs, o, 32);
  float rs = rsqrtf(vs * (1.0f / (float)T_D) + 1e-5f);
  int allz = 1;
#pragma unroll
  for (int i = 0; i < 10; i++) {
    int c = lane + i * 32;
    if (c < T_D) {
      float o = (vals[i] - mean) * rs * lns[c] + lnb[c];
      hout[(size_t)mrow * T_D + c] = o;
      h16[(size_t)mrow * T_DP + c] = (half_t)o;
      allz &= (o == 0.0f) ? 1 : 0;
    } else if (c < T_DP) {
      h16[(size_t)mrow * T_DP + c] = (half_t)0.0f;
    }
  }
  for (int o = 16; o > 0; o >>= 1) allz &= __shfl_xor(allz, o, 32);
  if (lane == 0) mask[mrow] = (unsigned char)(allz ? 1 : 0);
}

// ---------------------------------------------------------------------------
// Output heads (tiny)
// ---------------------------------------------------------------------------
__global__ void k_head1(const float* __restrict__ h, const float* __restrict__ w,
                        const float* __restrict__ b, float* __restrict__ sbuf) {
  int m = blockIdx.x * blockDim.x + threadIdx.x;
  if (m >= T_M) return;
  const float* row = h + (size_t)m * T_D;
  float acc = 0.0f;
  for (int c = 0; c < T_D; c++) acc += row[c] * w[c];
  sbuf[m] = acc + b[0];
}

__global__ void k_head2(const float* __restrict__ sbuf, const float* __restrict__ w,
                        const float* __restrict__ b, float* __restrict__ t2) {
  int idx = blockIdx.x * blockDim.x + threadIdx.x;
  if (idx >= T_B * 50) return;
  int j = idx % 50, bb = idx / 50;
  float acc = 0.0f;
  for (int s = 0; s < T_S; s++) acc += sbuf[bb * T_S + s] * w[s * 50 + j];
  t2[idx] = gelu_f(acc + b[j]);
}

__global__ void k_head3(const float* __restrict__ t2, const float* __restrict__ lw,
                        const float* __restrict__ lb, const float* __restrict__ aw,
                        const float* __restrict__ ab, float* __restrict__ out) {
  int idx = blockIdx.x * blockDim.x + threadIdx.x;
  if (idx >= T_B * 7) return;
  int o = idx % 7, bb = idx / 7;
  float acc = 0.0f;
  if (o == 0) {
    for (int i = 0; i < 50; i++) acc += t2[bb * 50 + i] * lw[i];
    acc += lb[0];
  } else {
    int j = o - 1;
    for (int i = 0; i < 50; i++) acc += t2[bb * 50 + i] * aw[i * 6 + j];
    acc += ab[j];
  }
  out[idx] = acc;
}

// ---------------------------------------------------------------------------
// Launch
// ---------------------------------------------------------------------------
extern "C" void kernel_launch(void* const* d_in, const int* in_sizes, int n_in,
                              void* d_out, int out_size, void* d_ws, size_t ws_size,
                              hipStream_t stream) {
  (void)in_sizes; (void)n_in; (void)out_size; (void)ws_size;
  const int*   x_tok  = (const int*)d_in[0];
  const float* emb    = (const float*)d_in[1];
  const float* qkv_w  = (const float*)d_in[2];
  const float* qkv_b  = (const float*)d_in[3];
  const float* out_w  = (const float*)d_in[4];
  const float* out_b  = (const float*)d_in[5];
  const float* ln1_s  = (const float*)d_in[6];
  const float* ln1_b  = (const float*)d_in[7];
  const float* ff1_w  = (const float*)d_in[8];
  const float* ff1_b  = (const float*)d_in[9];
  const float* ff2_w  = (const float*)d_in[10];
  const float* ff2_b  = (const float*)d_in[11];
  const float* ln2_s  = (const float*)d_in[12];
  const float* ln2_b  = (const float*)d_in[13];
  const float* lin1_w = (const float*)d_in[14];
  const float* lin1_b = (const float*)d_in[15];
  const float* lin2_w = (const float*)d_in[16];
  const float* lin2_b = (const float*)d_in[17];
  const float* lout_w = (const float*)d_in[18];
  const float* lout_b = (const float*)d_in[19];
  const float* laux_w = (const float*)d_in[20];
  const float* laux_b = (const float*)d_in[21];
  float* out = (float*)d_out;

  char* base = (char*)d_ws;
  size_t off = 0;
  auto alloc = [&](size_t bytes) -> void* {
    void* p = base + off;
    off = (off + bytes + 255) & ~(size_t)255;
    return p;
  };
  float*         h      = (float*)alloc((size_t)T_M * T_D * 4);
  half_t*        h16    = (half_t*)alloc((size_t)T_M * T_DP * 2);
  unsigned char* mask   = (unsigned char*)alloc(T_M);
  unsigned char* allpad = (unsigned char*)alloc(T_B);
  half_t*        wq     = (half_t*)alloc((size_t)5 * 320 * 960 * 2);  // [960][320] N-major
  half_t*        wo     = (half_t*)alloc((size_t)5 * 320 * 320 * 2);
  half_t*        wf1    = (half_t*)alloc((size_t)5 * 320 * 640 * 2);
  half_t*        wf2    = (half_t*)alloc((size_t)5 * 640 * 320 * 2);
  half_t*        qkv16  = (half_t*)alloc((size_t)T_M * 960 * 2);  // aliased as ff1out
  half_t*        qp     = (half_t*)alloc((size_t)T_B * T_H * T_SP * 64 * 2);
  half_t*        kp     = (half_t*)alloc((size_t)T_B * T_H * T_SP * 64 * 2);
  half_t*        vtp    = (half_t*)alloc((size_t)T_B * T_H * 64 * T_SP * 2);
  half_t*        attn16 = (half_t*)alloc((size_t)T_M * T_DP * 2);
  float*         c1     = (float*)alloc((size_t)T_M * T_DP * 4);
  float*         sbuf   = (float*)alloc((size_t)T_M * 4);
  float*         t2     = (float*)alloc((size_t)T_B * 50 * 4);
  half_t*        ff1out = qkv16;  // temporally disjoint use, [M][640] fits in [M][960]

  k_pad_w<<<(5 * 320 * 960 + 255) / 256, 256, 0, stream>>>(qkv_w, wq, 300, 900, 320, 960, 5);
  k_pad_w<<<(5 * 320 * 320 + 255) / 256, 256, 0, stream>>>(out_w, wo, 300, 300, 320, 320, 5);
  k_pad_w<<<(5 * 320 * 640 + 255) / 256, 256, 0, stream>>>(ff1_w, wf1, 300, 600, 320, 640, 5);
  k_pad_w<<<(5 * 640 * 320 + 255) / 256, 256, 0, stream>>>(ff2_w, wf2, 600, 300, 640, 320, 5);
  k_zero_attn_pad<<<(T_M * (T_DP - T_D) + 255) / 256, 256, 0, stream>>>(attn16);
  k_embed<<<(T_M + 255) / 256, 256, 0, stream>>>(x_tok, emb, h, h16, mask);
  k_allpad<<<1, 256, 0, stream>>>(mask, allpad);

  const size_t gemm_lds = 8192;
  const size_t attn_lds = (size_t)T_SP * 64 * 2 * 2 + (size_t)64 * T_SP * 4; // 114688

  for (int l = 0; l < 5; l++) {
    k_gemm<<<dim3(960 / 64, T_M / 64), 128, gemm_lds, stream>>>(
        h16, wq + (size_t)l * 320 * 960, qkv_b + l * 900, 900, 320, 960,
        nullptr, qkv16, 0);
    k_qkv_pack<<<(T_B * T_H * T_SP * 64 + 255) / 256, 256, 0, stream>>>(qkv16, qp, kp, vtp);
    k_attn<<<dim3(4, T_H, T_B), 128, attn_lds, stream>>>(qp, kp, vtp, mask, allpad, attn16);
    k_gemm<<<dim3(320 / 64, T_M / 64), 128, gemm_lds, stream>>>(
        attn16, wo + (size_t)l * 320 * 320, out_b + l * 300, 300, 320, 320,
        c1, nullptr, 0);
    k_add_ln<<<T_M, 32, 0, stream>>>(h, c1, ln1_s + l * 300, ln1_b + l * 300, h, h16, mask);
    k_gemm<<<dim3(640 / 64, T_M / 64), 128, gemm_lds, stream>>>(
        h16, wf1 + (size_t)l * 320 * 640, ff1_b + l * 600, 600, 320, 640,
        nullptr, ff1out, 1);
    k_gemm<<<dim3(320 / 64, T_M / 64), 128, gemm_lds, stream>>>(
        ff1out, wf2 + (size_t)l * 640 * 320, ff2_b + l * 300, 300, 640, 320,
        c1, nullptr, 0);
    k_add_ln<<<T_M, 32, 0, stream>>>(h, c1, ln2_s + l * 300, ln2_b + l * 300, h, h16, mask);
    k_allpad<<<1, 256, 0, stream>>>(mask, allpad);
  }

  k_head1<<<(T_M + 255) / 256, 256, 0, stream>>>(h, lin1_w, lin1_b, sbuf);
  k_head2<<<(T_B * 50 + 255) / 256, 256, 0, stream>>>(sbuf, lin2_w, lin2_b, t2);
  k_head3<<<(T_B * 7 + 255) / 256, 256, 0, stream>>>(t2, lout_w, lout_b, laux_w, laux_b, out);
}
